// RNN_20383914787559
// MI455X (gfx1250) — compile-verified, matrix operand order checked by reading, not run on hardware
//
#include <hip/hip_runtime.h>

typedef _Float16 f16;
typedef __attribute__((ext_vector_type(16))) _Float16 v16h;
typedef __attribute__((ext_vector_type(8)))  float    v8f;
typedef __attribute__((ext_vector_type(8)))  unsigned v8u;

#define TLEN  4096
#define BATCH 64
#define CIN   64
#define H     96
#define NTHR  192
#define XELEM (16 * CIN)   // 1024 x-values staged per step per WG
#define XPER  6            // ceil(1024 / 192)

// pack two floats into one dword of f16 pairs (A/B fragment element)
__device__ __forceinline__ unsigned pack2f(float a, float b) {
  f16 ha = (f16)a, hb = (f16)b;
  unsigned short ua = __builtin_bit_cast(unsigned short, ha);
  unsigned short ub = __builtin_bit_cast(unsigned short, hb);
  return (unsigned)ua | ((unsigned)ub << 16);
}

// fast gate math: v_exp_f32 + v_rcp_f32 (trans ops co-execute with WMMA),
// no IEEE division sequences on the recurrent critical path
__device__ __forceinline__ float sigmoid_fast(float x) {
  return __builtin_amdgcn_rcpf(1.0f + __expf(-x));
}
__device__ __forceinline__ float tanh_fast(float x) {
  // tanh(x) = 1 - 2/(e^{2x}+1)
  float e = __expf(2.0f * x);
  return __builtin_fmaf(-2.0f, __builtin_amdgcn_rcpf(e + 1.0f), 1.0f);
}

// K index within a 16x16x32 f16 fragment for VGPR r, lane-half lh
// (CDNA5 ISA 7.12.2: V0..3 hold K=0..7 / 8..15 by lane half, V4..7 hold +16)
__device__ __forceinline__ int kloc(int r, int lh) {
  return (r < 4) ? (lh * 8 + 2 * r) : (16 + lh * 8 + 2 * (r - 4));
}

__global__ __launch_bounds__(NTHR, 1)
void lstm_fused_kernel(const float* __restrict__ x,
                       const float* __restrict__ W_ih,
                       const float* __restrict__ W_hh,
                       const float* __restrict__ b_ih,
                       const float* __restrict__ b_hh,
                       float* __restrict__ out) {
  __shared__ __align__(16) f16 hbuf[16 * H];    // h tile   [m][k], f16
  __shared__ __align__(16) f16 xbuf[XELEM];     // x_t tile [m][c], f16

  const int tid   = threadIdx.x;
  const int wave  = tid >> 5;      // 0..5  -> 16-wide N block within each gate
  const int lane  = tid & 31;
  const int l16   = lane & 15;
  const int lh    = lane >> 4;     // 0/1
  const int bbase = blockIdx.x * 16;

  // ---------- preload weights as WMMA B-tiles into registers ----------
  v16h Bhh[4][3];   // 4 gates x K-tiles of 32 over H=96
  v16h Bih[4][2];   // 4 gates x K-tiles of 32 over CIN=64
  v8f  bias_acc[4]; // bias broadcast along M, used as C of the first WMMA
  #pragma unroll
  for (int q = 0; q < 4; ++q) {
    const int n = q * H + wave * 16 + l16;     // global gate row 0..383
    const float bq = b_ih[n] + b_hh[n];
    #pragma unroll
    for (int r = 0; r < 8; ++r) bias_acc[q][r] = bq;
    #pragma unroll
    for (int kt = 0; kt < 3; ++kt) {
      v8u tmp;
      #pragma unroll
      for (int r = 0; r < 8; ++r) {
        const int k0 = kt * 32 + kloc(r, lh);
        tmp[r] = pack2f(W_hh[n * H + k0], W_hh[n * H + k0 + 1]);
      }
      Bhh[q][kt] = __builtin_bit_cast(v16h, tmp);
    }
    #pragma unroll
    for (int kt = 0; kt < 2; ++kt) {
      v8u tmp;
      #pragma unroll
      for (int r = 0; r < 8; ++r) {
        const int k0 = kt * 32 + kloc(r, lh);
        tmp[r] = pack2f(W_ih[n * CIN + k0], W_ih[n * CIN + k0 + 1]);
      }
      Bih[q][kt] = __builtin_bit_cast(v16h, tmp);
    }
  }

  // ---------- init state: h = 0 (LDS), c = 0 (registers, per-wave tile) ----------
  for (int i = tid; i < 16 * H; i += NTHR) hbuf[i] = (f16)0.0f;
  v8f cc;
  #pragma unroll
  for (int r = 0; r < 8; ++r) cc[r] = 0.0f;

  // ---------- software pipeline: x(t) held in registers one step ahead ----------
  float xr[XPER];
  #pragma unroll
  for (int i = 0; i < XPER; ++i) {
    const int idx = tid + i * NTHR;
    if (idx < XELEM) {
      const int m = idx >> 6, c = idx & 63;
      xr[i] = x[((size_t)(bbase + m) * CIN + c) * TLEN + 0];
    }
  }
  __syncthreads();

  const int ncol = wave * 16 + l16;            // h column 0..95 owned by this lane

  #pragma unroll 1
  for (int t = 0; t < TLEN; ++t) {
    // ---- stage prefetched x registers -> LDS as f16 (no global latency here) ----
    #pragma unroll
    for (int i = 0; i < XPER; ++i) {
      const int idx = tid + i * NTHR;
      if (idx < XELEM) xbuf[idx] = (f16)xr[i];
    }
    __syncthreads();

    // ---- issue global loads for x(t+1); latency hidden under WMMA + gates ----
    if (t + 1 < TLEN) {
      #pragma unroll
      for (int i = 0; i < XPER; ++i) {
        const int idx = tid + i * NTHR;
        if (idx < XELEM) {
          const int m = idx >> 6, c = idx & 63;
          const float* p = &x[((size_t)(bbase + m) * CIN + c) * TLEN + (t + 1)];
          xr[i] = *p;
          __builtin_prefetch(p + 16, 0, 0);    // global_prefetch_b8: warm next line
        }
      }
    }

    // ---- input projection: K = 64 over x; first WMMA consumes bias as C ----
    v8f acc[4];
    #pragma unroll
    for (int kt = 0; kt < 2; ++kt) {
      v8u tmp;
      #pragma unroll
      for (int r = 0; r < 8; ++r) {
        const int k0 = kt * 32 + kloc(r, lh);
        tmp[r] = *(const unsigned*)&xbuf[l16 * CIN + k0];   // merges to ds_load_b128
      }
      v16h A = __builtin_bit_cast(v16h, tmp);
      #pragma unroll
      for (int q = 0; q < 4; ++q) {
        const v8f cin = (kt == 0) ? bias_acc[q] : acc[q];
        acc[q] = __builtin_amdgcn_wmma_f32_16x16x32_f16(
            false, A, false, Bih[q][kt], (short)0, cin, false, false);
      }
    }

    // ---- recurrent projection: K = 96 over h ----
    #pragma unroll
    for (int kt = 0; kt < 3; ++kt) {
      v8u tmp;
      #pragma unroll
      for (int r = 0; r < 8; ++r) {
        const int k0 = kt * 32 + kloc(r, lh);
        tmp[r] = *(const unsigned*)&hbuf[l16 * H + k0];
      }
      v16h A = __builtin_bit_cast(v16h, tmp);
      #pragma unroll
      for (int q = 0; q < 4; ++q)
        acc[q] = __builtin_amdgcn_wmma_f32_16x16x32_f16(
            false, A, false, Bhh[q][kt], (short)0, acc[q], false, false);
    }
    __syncthreads();   // all LDS reads of h/x done before anyone overwrites

    // ---- gates + state update in registers; h -> LDS (f16) and out (f32) ----
    #pragma unroll
    for (int r = 0; r < 8; ++r) {
      const int m  = r + lh * 8;               // batch row within tile
      const float ig = sigmoid_fast(acc[0][r]);
      const float fg = sigmoid_fast(acc[1][r]);
      const float gg = tanh_fast(acc[2][r]);
      const float og = sigmoid_fast(acc[3][r]);
      const float cn = __builtin_fmaf(fg, cc[r], ig * gg);
      cc[r] = cn;
      const float hn = og * tanh_fast(cn);
      hbuf[m * H + ncol] = (f16)hn;
      out[((size_t)(bbase + m) * H + ncol) * TLEN + t] = hn;
    }
    // no barrier needed here: the next iteration's LDS writes/reads are fenced
    // by the __syncthreads() after the stage, and these writes precede it
  }
}

extern "C" void kernel_launch(void* const* d_in, const int* in_sizes, int n_in,
                              void* d_out, int out_size, void* d_ws, size_t ws_size,
                              hipStream_t stream) {
  (void)in_sizes; (void)n_in; (void)d_ws; (void)ws_size; (void)out_size;
  const float* x    = (const float*)d_in[0];
  const float* W_ih = (const float*)d_in[1];
  const float* W_hh = (const float*)d_in[2];
  const float* b_ih = (const float*)d_in[3];
  const float* b_hh = (const float*)d_in[4];
  float* out = (float*)d_out;

  // 4 workgroups x 16 batch rows; 6 waves (192 threads) cover N = 4 gates x 96
  lstm_fused_kernel<<<dim3(4), dim3(NTHR), 0, stream>>>(x, W_ih, W_hh, b_ih, b_hh, out);
}
